// TransformerBlock_60911226192506
// MI455X (gfx1250) — matrix operand order repaired and checked by reading, not compiled, not run on hardware
//
#include <hip/hip_runtime.h>

#define N_TOK 16384
#define M_TOK 8192
#define DDIM  128
#define HDIM  512
#define LDK   136   // 128 + 8 bf16 elements pad (16B) -> conflict-free b128 LDS loads
#define LDA2  520   // 512 + 8

#define USE_ASYNC_LDS 1

typedef __bf16 v16bf __attribute__((ext_vector_type(16)));
typedef float  v8f   __attribute__((ext_vector_type(8)));

static __device__ __forceinline__ float bf2f(unsigned short h){
  union { unsigned u; float f; } c; c.u = ((unsigned)h) << 16; return c.f;
}
static __device__ __forceinline__ unsigned short f2bf(float f){
  union { float f; unsigned u; } c; c.f = f;
  unsigned r = c.u + 0x7FFFu + ((c.u >> 16) & 1u);   // round-to-nearest-even
  return (unsigned short)(r >> 16);
}

static __device__ __forceinline__ v8f wmma_bf16(v16bf a, v16bf b, v8f c){
  return __builtin_amdgcn_wmma_f32_16x16x32_bf16(false, a, false, b, (short)0, c, false, false);
}

// ---- CDNA5 async global->LDS copy (ASYNCcnt path, no VGPR payload) ----
static __device__ __forceinline__ void async_copy_b128(unsigned ldsByteAddr, const void* g){
  asm volatile("global_load_async_to_lds_b128 %0, %1, off"
               :: "v"(ldsByteAddr), "v"(g) : "memory");
}
static __device__ __forceinline__ void async_wait0(){
  asm volatile("s_wait_asynccnt 0x0" ::: "memory");
}

// Stage nChunks x 16B from row-major global (srcRowBytes per row) into padded LDS rows.
template<int LOG2_CPR>
static __device__ __forceinline__ void stage_g2l(unsigned dstB, int ldsStride,
                                                 const char* src, int srcRowBytes,
                                                 int totalChunks, int tid,
                                                 unsigned char* ldsBytes){
#if USE_ASYNC_LDS
  (void)ldsBytes;
  for(int c = tid; c < totalChunks; c += 256){
    int r = c >> LOG2_CPR, k = c & ((1 << LOG2_CPR) - 1);
    async_copy_b128(dstB + r * ldsStride + k * 16, src + (size_t)r * srcRowBytes + k * 16);
  }
#else
  (void)dstB;
  for(int c = tid; c < totalChunks; c += 256){
    int r = c >> LOG2_CPR, k = c & ((1 << LOG2_CPR) - 1);
    *(uint4*)(ldsBytes + r * ldsStride + k * 16) =
        *(const uint4*)(src + (size_t)r * srcRowBytes + k * 16);
  }
#endif
}

union FragU { v16bf v; uint4 q[2]; };

// A-matrix 16x32 bf16 fragment from row-major LDS tile [m][k], stride ld (elements).
static __device__ __forceinline__ v16bf load_a_frag(const unsigned short* t, int ld,
                                                    int m, int halfSel, int kbase){
  FragU u;
  const unsigned short* p = t + m * ld + kbase + halfSel * 8;
  u.q[0] = *(const uint4*)(p);
  u.q[1] = *(const uint4*)(p + 16);
  return u.v;
}
// B-matrix 32x16 bf16 fragment from transposed LDS tile BT[n][k] (k contiguous).
static __device__ __forceinline__ v16bf load_b_frag(const unsigned short* t, int ld,
                                                    int n, int halfSel, int kbase){
  FragU u;
  const unsigned short* p = t + n * ld + kbase + halfSel * 8;
  u.q[0] = *(const uint4*)(p);
  u.q[1] = *(const uint4*)(p + 16);
  return u.v;
}

static __device__ __forceinline__ float sum8_bf(uint4 q){
  float s = 0.f;
  s += bf2f((unsigned short)(q.x & 0xFFFFu)) + bf2f((unsigned short)(q.x >> 16));
  s += bf2f((unsigned short)(q.y & 0xFFFFu)) + bf2f((unsigned short)(q.y >> 16));
  s += bf2f((unsigned short)(q.z & 0xFFFFu)) + bf2f((unsigned short)(q.z >> 16));
  s += bf2f((unsigned short)(q.w & 0xFFFFu)) + bf2f((unsigned short)(q.w >> 16));
  return s;
}

// ---------------------------------------------------------------- BN stats
__global__ void k_bnstats(const float* __restrict__ y, const float* __restrict__ g,
                          const float* __restrict__ b, float* __restrict__ scale,
                          float* __restrict__ shift){
  __shared__ float rs[256], rq[256];
  int c = blockIdx.x, t = threadIdx.x;
  float s = 0.f, s2 = 0.f;
  for(int r = t; r < M_TOK; r += 256){
    float v = y[(size_t)r * DDIM + c];
    s += v; s2 += v * v;
  }
  rs[t] = s; rq[t] = s2; __syncthreads();
  for(int o = 128; o > 0; o >>= 1){
    if(t < o){ rs[t] += rs[t + o]; rq[t] += rq[t + o]; }
    __syncthreads();
  }
  if(t == 0){
    float mu  = rs[0] * (1.f / M_TOK);
    float var = rq[0] * (1.f / M_TOK) - mu * mu;
    float sc  = g[c] * rsqrtf(var + 1e-5f);
    scale[c] = sc;
    shift[c] = b[c] - mu * sc;
  }
}

// ------------------------------------------------- LN(x) @ Wq + bq -> XQ (bf16)
__global__ void k_lnq(const float* __restrict__ x, const float* __restrict__ Wq,
                      const float* __restrict__ bq, const float* __restrict__ ln_g,
                      const float* __restrict__ ln_b, unsigned short* __restrict__ XQ){
  extern __shared__ unsigned char smem[];
  unsigned short* Zt = (unsigned short*)smem;        // [128][LDK]
  unsigned short* WT = Zt + 128 * LDK;               // [128][LDK] = Wq^T
  float* gamma = (float*)(WT + 128 * LDK);
  float* beta  = gamma + 128;
  float* bqs   = beta + 128;
  int tid = threadIdx.x;
  int rowBase = blockIdx.x * 128;

  if(tid < 128){ gamma[tid] = ln_g[tid]; beta[tid] = ln_b[tid]; bqs[tid] = bq[tid]; }
  for(int i = tid; i < DDIM * DDIM; i += 256){
    int k = i >> 7, n = i & 127;
    WT[n * LDK + k] = f2bf(Wq[i]);
  }
  __syncthreads();

  if(tid < 128){
    const float4* xr = (const float4*)(x + (size_t)(rowBase + tid) * DDIM);
    float s = 0.f, s2 = 0.f;
#pragma unroll
    for(int i = 0; i < 32; i++){
      float4 v = xr[i];
      s  += v.x + v.y + v.z + v.w;
      s2 += v.x * v.x + v.y * v.y + v.z * v.z + v.w * v.w;
    }
    float mu = s * (1.f / 128.f);
    float rstd = rsqrtf(s2 * (1.f / 128.f) - mu * mu + 1e-5f);
    unsigned short* zr = Zt + tid * LDK;
#pragma unroll
    for(int i = 0; i < 32; i++){
      float4 v = xr[i]; int c = i * 4;
      zr[c + 0] = f2bf((v.x - mu) * rstd * gamma[c + 0] + beta[c + 0]);
      zr[c + 1] = f2bf((v.y - mu) * rstd * gamma[c + 1] + beta[c + 1]);
      zr[c + 2] = f2bf((v.z - mu) * rstd * gamma[c + 2] + beta[c + 2]);
      zr[c + 3] = f2bf((v.w - mu) * rstd * gamma[c + 3] + beta[c + 3]);
    }
  }
  __syncthreads();

  int wave = tid >> 5, lane = tid & 31, half = lane >> 4, ln = lane & 15;
  v16bf aQ[4];
#pragma unroll
  for(int kc = 0; kc < 4; kc++) aQ[kc] = load_a_frag(Zt, LDK, wave * 16 + ln, half, kc * 32);
#pragma unroll
  for(int n0 = 0; n0 < 8; n0++){
    v8f acc = {0, 0, 0, 0, 0, 0, 0, 0};
#pragma unroll
    for(int kc = 0; kc < 4; kc++){
      v16bf b = load_b_frag(WT, LDK, n0 * 16 + ln, half, kc * 32);
      acc = wmma_bf16(aQ[kc], b, acc);
    }
    int n = n0 * 16 + ln;
    float bias = bqs[n];
#pragma unroll
    for(int r = 0; r < 8; r++){
      int m = wave * 16 + r + half * 8;
      XQ[(size_t)(rowBase + m) * DDIM + n] = f2bf(acc[r] + bias);
    }
  }
}

// ------------------------------- BN(y) @ {Wk,Wv} + {bk,bv} -> YK,YV (bf16)
__global__ void k_kv(const float* __restrict__ y, const float* __restrict__ Wk,
                     const float* __restrict__ bk, const float* __restrict__ Wv,
                     const float* __restrict__ bv, const float* __restrict__ scale,
                     const float* __restrict__ shift, unsigned short* __restrict__ YK,
                     unsigned short* __restrict__ YV){
  extern __shared__ unsigned char smem[];
  unsigned short* Yt  = (unsigned short*)smem;       // [128][LDK]
  unsigned short* WkT = Yt  + 128 * LDK;
  unsigned short* WvT = WkT + 128 * LDK;
  float* sc  = (float*)(WvT + 128 * LDK);
  float* sh  = sc + 128;
  float* bks = sh + 128;
  float* bvs = bks + 128;
  int tid = threadIdx.x;
  int rowBase = blockIdx.x * 128;

  if(tid < 128){ sc[tid] = scale[tid]; sh[tid] = shift[tid]; bks[tid] = bk[tid]; bvs[tid] = bv[tid]; }
  for(int i = tid; i < DDIM * DDIM; i += 256){
    int k = i >> 7, n = i & 127;
    WkT[n * LDK + k] = f2bf(Wk[i]);
    WvT[n * LDK + k] = f2bf(Wv[i]);
  }
  __syncthreads();
  for(int i = tid; i < 128 * DDIM; i += 256){
    int r = i >> 7, c = i & 127;
    float v = y[(size_t)rowBase * DDIM + i];
    Yt[r * LDK + c] = f2bf(v * sc[c] + sh[c]);
  }
  __syncthreads();

  int wave = tid >> 5, lane = tid & 31, half = lane >> 4, ln = lane & 15;
  v16bf aY[4];
#pragma unroll
  for(int kc = 0; kc < 4; kc++) aY[kc] = load_a_frag(Yt, LDK, wave * 16 + ln, half, kc * 32);
#pragma unroll
  for(int n0 = 0; n0 < 8; n0++){
    v8f accK = {0, 0, 0, 0, 0, 0, 0, 0};
    v8f accV = {0, 0, 0, 0, 0, 0, 0, 0};
#pragma unroll
    for(int kc = 0; kc < 4; kc++){
      v16bf bK = load_b_frag(WkT, LDK, n0 * 16 + ln, half, kc * 32);
      v16bf bV = load_b_frag(WvT, LDK, n0 * 16 + ln, half, kc * 32);
      accK = wmma_bf16(aY[kc], bK, accK);
      accV = wmma_bf16(aY[kc], bV, accV);
    }
    int n = n0 * 16 + ln;
    float biK = bks[n], biV = bvs[n];
#pragma unroll
    for(int r = 0; r < 8; r++){
      int m = wave * 16 + r + half * 8;
      size_t gi = (size_t)(rowBase + m) * DDIM + n;
      YK[gi] = f2bf(accK[r] + biK);
      YV[gi] = f2bf(accV[r] + biV);
    }
  }
}

// ---------------------- fused smooth-softmax attention: Z1 = A @ YV + x
__global__ void k_attn(const unsigned short* __restrict__ XQ,
                       const unsigned short* __restrict__ YK,
                       const unsigned short* __restrict__ YV,
                       const float* __restrict__ x, float* __restrict__ Z1){
  extern __shared__ unsigned char smem[];
  unsigned short* Qt = (unsigned short*)smem;        // [128][LDK]
  unsigned short* Kt = Qt + 128 * LDK;
  unsigned short* VT = Kt + 128 * LDK;               // [d][key]
  unsigned short* Pe = VT + 128 * LDK;
  unsigned short* Pr = Pe + 128 * LDK;
  float* invZ = (float*)(Pr + 128 * LDK);
  float* invD = invZ + 128;

  int tid = threadIdx.x;
  int wave = tid >> 5, lane = tid & 31, half = lane >> 4, ln = lane & 15;
  int rowBase = blockIdx.x * 128;

  unsigned QtB = (unsigned)(size_t)Qt;   // low 32 bits of flat LDS addr == LDS byte offset
  unsigned KtB = (unsigned)(size_t)Kt;

  // stage Q tile (async, 128 rows x 16 chunks of 16B)
  stage_g2l<4>(QtB, LDK * 2, (const char*)(XQ + (size_t)rowBase * DDIM), DDIM * 2,
               2048, tid, (unsigned char*)Qt);
#if USE_ASYNC_LDS
  async_wait0();
#endif
  __syncthreads();

  v16bf aQ[4];
#pragma unroll
  for(int kc = 0; kc < 4; kc++) aQ[kc] = load_a_frag(Qt, LDK, wave * 16 + ln, half, kc * 32);

  v8f accE[8], accR[8];
#pragma unroll
  for(int n0 = 0; n0 < 8; n0++){
    accE[n0] = (v8f){0, 0, 0, 0, 0, 0, 0, 0};
    accR[n0] = (v8f){0, 0, 0, 0, 0, 0, 0, 0};
  }
  float Zacc = 0.f, Racc = 0.f;

  for(int jt = 0; jt < M_TOK / 128; jt++){
    // K tile: async copy, layout-preserving rows
    stage_g2l<4>(KtB, LDK * 2, (const char*)(YK + (size_t)jt * 128 * DDIM), DDIM * 2,
                 2048, tid, (unsigned char*)Kt);
    // V tile: transpose into VT[d][key] (b128 global loads -> ds b16 stores)
    {
      const unsigned short* vsrc = YV + (size_t)jt * 128 * DDIM;
      for(int c = tid; c < 2048; c += 256){
        int key = c >> 4, d8 = (c & 15) * 8;
        uint4 q = *(const uint4*)(vsrc + (size_t)key * DDIM + d8);
        VT[(d8 + 0) * LDK + key] = (unsigned short)(q.x & 0xFFFFu);
        VT[(d8 + 1) * LDK + key] = (unsigned short)(q.x >> 16);
        VT[(d8 + 2) * LDK + key] = (unsigned short)(q.y & 0xFFFFu);
        VT[(d8 + 3) * LDK + key] = (unsigned short)(q.y >> 16);
        VT[(d8 + 4) * LDK + key] = (unsigned short)(q.z & 0xFFFFu);
        VT[(d8 + 5) * LDK + key] = (unsigned short)(q.z >> 16);
        VT[(d8 + 6) * LDK + key] = (unsigned short)(q.w & 0xFFFFu);
        VT[(d8 + 7) * LDK + key] = (unsigned short)(q.w >> 16);
      }
    }
#if USE_ASYNC_LDS
    async_wait0();
#endif
    __syncthreads();

    // S = Q @ K^T ; write exp(S-32), relu(S) as bf16 P-tiles
#pragma unroll
    for(int n0 = 0; n0 < 8; n0++){
      v8f s = {0, 0, 0, 0, 0, 0, 0, 0};
#pragma unroll
      for(int kc = 0; kc < 4; kc++){
        v16bf b = load_b_frag(Kt, LDK, n0 * 16 + ln, half, kc * 32);
        s = wmma_bf16(aQ[kc], b, s);
      }
      int col = n0 * 16 + ln;
#pragma unroll
      for(int r = 0; r < 8; r++){
        int m = wave * 16 + r + half * 8;
        float sv = s[r];
        Pe[m * LDK + col] = f2bf(__expf(sv - 32.0f));
        Pr[m * LDK + col] = f2bf(fmaxf(sv, 0.0f));
      }
    }
    __syncthreads();

    // running row sums (thread t owns row t)
    if(tid < 128){
      const uint4* pe = (const uint4*)(Pe + tid * LDK);
      const uint4* pr = (const uint4*)(Pr + tid * LDK);
      float se = 0.f, sr = 0.f;
#pragma unroll
      for(int q = 0; q < 16; q++){ se += sum8_bf(pe[q]); sr += sum8_bf(pr[q]); }
      Zacc += se; Racc += sr;
    }

    // accE += Pexp @ V ; accR += Prelu @ V
#pragma unroll
    for(int kc = 0; kc < 4; kc++){
      v16bf aE = load_a_frag(Pe, LDK, wave * 16 + ln, half, kc * 32);
      v16bf aR = load_a_frag(Pr, LDK, wave * 16 + ln, half, kc * 32);
#pragma unroll
      for(int n0 = 0; n0 < 8; n0++){
        v16bf bV = load_b_frag(VT, LDK, n0 * 16 + ln, half, kc * 32);
        accE[n0] = wmma_bf16(aE, bV, accE[n0]);
        accR[n0] = wmma_bf16(aR, bV, accR[n0]);
      }
    }
    __syncthreads();
  }

  if(tid < 128){
    invZ[tid] = 1.0f / (Zacc + 1e-30f);
    invD[tid] = 1.0f / (0.1f * Racc + 1.0f);
  }
  __syncthreads();

  float iz[8], idn[8];
#pragma unroll
  for(int r = 0; r < 8; r++){
    int m = wave * 16 + r + half * 8;
    iz[r] = invZ[m]; idn[r] = invD[m];
  }
#pragma unroll
  for(int n0 = 0; n0 < 8; n0++){
    int n = n0 * 16 + ln;
#pragma unroll
    for(int r = 0; r < 8; r++){
      int m = wave * 16 + r + half * 8;
      size_t gi = (size_t)(rowBase + m) * DDIM + n;
      Z1[gi] = (0.1f * accR[n0][r] + accE[n0][r] * iz[r]) * idn[r] + x[gi];
    }
  }
}

// --------------------------- H1 = relu(LN(Z1) @ W1 + b1)  (bf16 out)
__global__ void k_mlp1(const float* __restrict__ Z1, const float* __restrict__ W1,
                       const float* __restrict__ b1, const float* __restrict__ ln_g,
                       const float* __restrict__ ln_b, unsigned short* __restrict__ H1){
  extern __shared__ unsigned char smem[];
  unsigned short* Zt = (unsigned short*)smem;        // [128][LDK]
  unsigned short* WT = Zt + 128 * LDK;               // [512][LDK] = W1^T
  float* gamma = (float*)(WT + HDIM * LDK);
  float* beta  = gamma + 128;
  float* b1s   = beta + 128;                         // 512 floats
  int tid = threadIdx.x;
  int rowBase = blockIdx.x * 128;

  if(tid < 128){ gamma[tid] = ln_g[tid]; beta[tid] = ln_b[tid]; }
  for(int i = tid; i < HDIM; i += 256) b1s[i] = b1[i];
  for(int i = tid; i < DDIM * HDIM; i += 256){
    int k = i >> 9, n = i & 511;
    WT[n * LDK + k] = f2bf(W1[i]);
  }
  __syncthreads();

  if(tid < 128){
    const float4* xr = (const float4*)(Z1 + (size_t)(rowBase + tid) * DDIM);
    float s = 0.f, s2 = 0.f;
#pragma unroll
    for(int i = 0; i < 32; i++){
      float4 v = xr[i];
      s  += v.x + v.y + v.z + v.w;
      s2 += v.x * v.x + v.y * v.y + v.z * v.z + v.w * v.w;
    }
    float mu = s * (1.f / 128.f);
    float rstd = rsqrtf(s2 * (1.f / 128.f) - mu * mu + 1e-5f);
    unsigned short* zr = Zt + tid * LDK;
#pragma unroll
    for(int i = 0; i < 32; i++){
      float4 v = xr[i]; int c = i * 4;
      zr[c + 0] = f2bf((v.x - mu) * rstd * gamma[c + 0] + beta[c + 0]);
      zr[c + 1] = f2bf((v.y - mu) * rstd * gamma[c + 1] + beta[c + 1]);
      zr[c + 2] = f2bf((v.z - mu) * rstd * gamma[c + 2] + beta[c + 2]);
      zr[c + 3] = f2bf((v.w - mu) * rstd * gamma[c + 3] + beta[c + 3]);
    }
  }
  __syncthreads();

  int wave = tid >> 5, lane = tid & 31, half = lane >> 4, ln = lane & 15;
  v16bf aZ[4];
#pragma unroll
  for(int kc = 0; kc < 4; kc++) aZ[kc] = load_a_frag(Zt, LDK, wave * 16 + ln, half, kc * 32);
  for(int n0 = 0; n0 < 32; n0++){
    v8f acc = {0, 0, 0, 0, 0, 0, 0, 0};
#pragma unroll
    for(int kc = 0; kc < 4; kc++){
      v16bf b = load_b_frag(WT, LDK, n0 * 16 + ln, half, kc * 32);
      acc = wmma_bf16(aZ[kc], b, acc);
    }
    int n = n0 * 16 + ln;
    float bias = b1s[n];
#pragma unroll
    for(int r = 0; r < 8; r++){
      int m = wave * 16 + r + half * 8;
      H1[(size_t)(rowBase + m) * HDIM + n] = f2bf(fmaxf(acc[r] + bias, 0.f));
    }
  }
}

// --------------------------- H2 = relu(H1 @ W2 + b2)  (bf16)
__global__ void k_mlp2(const unsigned short* __restrict__ H1, const float* __restrict__ W2,
                       const float* __restrict__ b2, unsigned short* __restrict__ H2){
  extern __shared__ unsigned char smem[];
  unsigned short* At = (unsigned short*)smem;        // [128][LDA2]
  unsigned short* WT = At + 128 * LDA2;              // [128 n][LDA2 k] tile of W2^T
  float* b2s = (float*)(WT + 128 * LDA2);            // 512 floats
  int tid = threadIdx.x;
  int rowBase = blockIdx.x * 128;

  for(int i = tid; i < HDIM; i += 256) b2s[i] = b2[i];
  stage_g2l<6>((unsigned)(size_t)At, LDA2 * 2, (const char*)(H1 + (size_t)rowBase * HDIM),
               HDIM * 2, 8192, tid, (unsigned char*)At);
#if USE_ASYNC_LDS
  async_wait0();
#endif
  __syncthreads();

  int wave = tid >> 5, lane = tid & 31, half = lane >> 4, ln = lane & 15;
  v16bf aF[16];
#pragma unroll
  for(int kc = 0; kc < 16; kc++) aF[kc] = load_a_frag(At, LDA2, wave * 16 + ln, half, kc * 32);

  for(int nb = 0; nb < 4; nb++){
    __syncthreads();
    for(int i = tid; i < HDIM * 128; i += 256){
      int k = i >> 7, n = i & 127;
      WT[n * LDA2 + k] = f2bf(W2[(size_t)k * HDIM + nb * 128 + n]);
    }
    __syncthreads();
#pragma unroll
    for(int n0 = 0; n0 < 8; n0++){
      v8f acc = {0, 0, 0, 0, 0, 0, 0, 0};
#pragma unroll
      for(int kc = 0; kc < 16; kc++){
        v16bf b = load_b_frag(WT, LDA2, n0 * 16 + ln, half, kc * 32);
        acc = wmma_bf16(aF[kc], b, acc);
      }
      int n = nb * 128 + n0 * 16 + ln;
      float bias = b2s[n];
#pragma unroll
      for(int r = 0; r < 8; r++){
        int m = wave * 16 + r + half * 8;
        H2[(size_t)(rowBase + m) * HDIM + n] = f2bf(fmaxf(acc[r] + bias, 0.f));
      }
    }
  }
}

// --------------------------- out = H2 @ W3 + b3 + Z1
__global__ void k_mlp3(const unsigned short* __restrict__ H2, const float* __restrict__ W3,
                       const float* __restrict__ b3, const float* __restrict__ Z1,
                       float* __restrict__ out){
  extern __shared__ unsigned char smem[];
  unsigned short* At = (unsigned short*)smem;        // [128][LDA2]
  unsigned short* WT = At + 128 * LDA2;              // [128 n][LDA2 k] = W3^T
  float* b3s = (float*)(WT + 128 * LDA2);            // 128 floats
  int tid = threadIdx.x;
  int rowBase = blockIdx.x * 128;

  if(tid < 128) b3s[tid] = b3[tid];
  stage_g2l<6>((unsigned)(size_t)At, LDA2 * 2, (const char*)(H2 + (size_t)rowBase * HDIM),
               HDIM * 2, 8192, tid, (unsigned char*)At);
  for(int i = tid; i < HDIM * DDIM; i += 256){
    int k = i >> 7, n = i & 127;
    WT[n * LDA2 + k] = f2bf(W3[i]);
  }
#if USE_ASYNC_LDS
  async_wait0();
#endif
  __syncthreads();

  int wave = tid >> 5, lane = tid & 31, half = lane >> 4, ln = lane & 15;
  v16bf aF[16];
#pragma unroll
  for(int kc = 0; kc < 16; kc++) aF[kc] = load_a_frag(At, LDA2, wave * 16 + ln, half, kc * 32);
#pragma unroll
  for(int n0 = 0; n0 < 8; n0++){
    v8f acc = {0, 0, 0, 0, 0, 0, 0, 0};
#pragma unroll
    for(int kc = 0; kc < 16; kc++){
      v16bf b = load_b_frag(WT, LDA2, n0 * 16 + ln, half, kc * 32);
      acc = wmma_bf16(aF[kc], b, acc);
    }
    int n = n0 * 16 + ln;
    float bias = b3s[n];
#pragma unroll
    for(int r = 0; r < 8; r++){
      int m = wave * 16 + r + half * 8;
      size_t gi = (size_t)(rowBase + m) * DDIM + n;
      out[gi] = acc[r] + bias + Z1[gi];
    }
  }
}

// ---------------------------------------------------------------- launch
extern "C" void kernel_launch(void* const* d_in, const int* in_sizes, int n_in,
                              void* d_out, int out_size, void* d_ws, size_t ws_size,
                              hipStream_t stream){
  const float* x    = (const float*)d_in[0];
  const float* y    = (const float*)d_in[1];
  const float* Wq   = (const float*)d_in[2];  const float* bq = (const float*)d_in[3];
  const float* Wk   = (const float*)d_in[4];  const float* bk = (const float*)d_in[5];
  const float* Wv   = (const float*)d_in[6];  const float* bv = (const float*)d_in[7];
  const float* ln_g = (const float*)d_in[8];  const float* ln_b = (const float*)d_in[9];
  const float* bn_g = (const float*)d_in[10]; const float* bn_b = (const float*)d_in[11];
  const float* W1   = (const float*)d_in[12]; const float* b1 = (const float*)d_in[13];
  const float* W2   = (const float*)d_in[14]; const float* b2 = (const float*)d_in[15];
  const float* W3   = (const float*)d_in[16]; const float* b3 = (const float*)d_in[17];
  float* out = (float*)d_out;

  char* ws = (char*)d_ws;
  float* bnScale = (float*)(ws);
  float* bnShift = (float*)(ws + 512);
  unsigned short* XQ = (unsigned short*)(ws + 1024);
  unsigned short* YK = XQ + (size_t)N_TOK * DDIM;
  unsigned short* YV = YK + (size_t)M_TOK * DDIM;
  float* Z1 = (float*)(YV + (size_t)M_TOK * DDIM);
  unsigned short* H1 = (unsigned short*)(Z1 + (size_t)N_TOK * DDIM);
  unsigned short* H2 = H1 + (size_t)N_TOK * HDIM;

  constexpr int SM_TILE = 128 * LDK * 2;                           // 34816 B
  constexpr int SM_LNQ  = SM_TILE * 2 + 2048;                      //  71680
  constexpr int SM_KV   = SM_TILE * 3 + 2048;                      // 106496
  constexpr int SM_ATTN = SM_TILE * 5 + 1024;                      // 175104
  constexpr int SM_M1   = SM_TILE + HDIM * LDK * 2 + 3072;         // 177152
  constexpr int SM_M2   = 128 * LDA2 * 2 * 2 + 2048;               // 268288
  constexpr int SM_M3   = 128 * LDA2 * 2 * 2 + 512;                // 266752

  k_bnstats<<<DDIM, 256, 0, stream>>>(y, bn_g, bn_b, bnScale, bnShift);
  k_lnq<<<N_TOK / 128, 256, SM_LNQ, stream>>>(x, Wq, bq, ln_g, ln_b, XQ);
  k_kv<<<M_TOK / 128, 256, SM_KV, stream>>>(y, Wk, bk, Wv, bv, bnScale, bnShift, YK, YV);
  k_attn<<<N_TOK / 128, 256, SM_ATTN, stream>>>(XQ, YK, YV, x, Z1);
  k_mlp1<<<N_TOK / 128, 256, SM_M1, stream>>>(Z1, W1, b1, ln_g, ln_b, H1);
  k_mlp2<<<N_TOK / 128, 256, SM_M2, stream>>>(H1, W2, b2, H2);
  k_mlp3<<<N_TOK / 128, 256, SM_M3, stream>>>(H2, W3, b3, Z1, out);
}
